// Attention_10471130268299
// MI455X (gfx1250) — matrix operand order repaired and checked by reading, not compiled, not run on hardware
//
#include <hip/hip_runtime.h>
#include <hip/hip_bf16.h>

typedef __bf16 bf16_t;
typedef __attribute__((ext_vector_type(16))) __bf16 v16bf;
typedef __attribute__((ext_vector_type(8)))  __bf16 v8bf;
typedef __attribute__((ext_vector_type(8)))  float  v8f;
typedef __attribute__((ext_vector_type(4)))  unsigned int u32x4;
typedef __attribute__((ext_vector_type(8)))  int i32x8;
typedef __attribute__((ext_vector_type(4)))  int i32x4;

#define B_    2
#define SEQ_  2048
#define DIM_  4096
#define NH_   32
#define NKV_  8
#define HD_   128
#define MTOT_ (B_ * SEQ_)   // 4096 token rows

#if defined(__has_builtin)
#if __has_builtin(__builtin_amdgcn_tensor_load_to_lds) && \
    __has_builtin(__builtin_amdgcn_s_wait_tensorcnt)
#define HAVE_TDM 1
#endif
#endif

static __device__ __forceinline__ v8f wmma_bf16(v16bf a, v16bf b, v8f c) {
  // D = A(16x32 bf16) * B(32x16 bf16) + C(16x16 f32)
  return __builtin_amdgcn_wmma_f32_16x16x32_bf16(false, a, false, b, (short)0, c,
                                                 false, false);
}

union BFV16 { v16bf v; v8bf h8[2]; };

static __device__ __forceinline__ v8bf load8(const float* p) {
  const float4 f0 = *(const float4*)p;
  const float4 f1 = *(const float4*)(p + 4);
  v8bf r;
  r[0] = (bf16_t)f0.x; r[1] = (bf16_t)f0.y; r[2] = (bf16_t)f0.z; r[3] = (bf16_t)f0.w;
  r[4] = (bf16_t)f1.x; r[5] = (bf16_t)f1.y; r[6] = (bf16_t)f1.z; r[7] = (bf16_t)f1.w;
  return r;
}

#ifdef HAVE_TDM
// LDS byte-offset of a __shared__ object (generic -> addrspace(3) -> int)
static __device__ __forceinline__ unsigned lds_addr_of(void* p) {
  return (unsigned)(unsigned long long)(__attribute__((address_space(3))) void*)p;
}

// Issue one TDM 2D tile load: rows x cols bf16, row stride = ldK elements,
// LDS destination padded to 40 elements/row via D# pad (interval 16dw, pad 4dw).
static __device__ __forceinline__ void tdm_load_tile_a(const bf16_t* gsrc,
                                                       unsigned lds_byte_off,
                                                       int rows, int cols,
                                                       int ldK, int tensor_m) {
  unsigned long long ga = (unsigned long long)(__UINTPTR_TYPE__)gsrc;
  u32x4 g0;
  g0[0] = 1u;                                   // count=1, user mode
  g0[1] = lds_byte_off;                         // lds_addr
  g0[2] = (unsigned)(ga & 0xffffffffu);         // global_addr[31:0]
  g0[3] = (unsigned)((ga >> 32) & 0x01ffffffu)  // global_addr[56:32]
          | (2u << 30);                         // type = 2 (image)
  unsigned td0 = (unsigned)ldK;                 // tensor_dim0 (elements)
  unsigned td1 = (unsigned)tensor_m;            // tensor_dim1
  unsigned long long st0 = (unsigned long long)ldK;  // tensor_dim0_stride
  i32x8 g1;
  g1[0] = (int)((1u << 16)        // data_size = 2 bytes
              | (1u << 20)        // pad_enable
              | (3u << 22)        // pad_interval: 16 DWORDs
              | (3u << 25));      // pad_amount:   4 DWORDs
  g1[1] = (int)((td0 & 0xffffu) << 16);                       // dim0[15:0] @63:48
  g1[2] = (int)(((td0 >> 16) & 0xffffu) | ((td1 & 0xffffu) << 16));
  g1[3] = (int)(((td1 >> 16) & 0xffffu) | ((unsigned)cols << 16)); // tile_dim0
  g1[4] = (int)((unsigned)rows & 0xffffu);                    // tile_dim1 (dim2=0)
  g1[5] = (int)(st0 & 0xffffffffu);
  g1[6] = (int)((st0 >> 32) & 0xffffu);                       // dim1_stride = 0
  g1[7] = 0;
  i32x4 z4 = {0, 0, 0, 0};
#if __clang_major__ >= 23
  i32x8 z8 = {0, 0, 0, 0, 0, 0, 0, 0};
  __builtin_amdgcn_tensor_load_to_lds(g0, g1, z4, z4, z8, 0);
#else
  __builtin_amdgcn_tensor_load_to_lds(g0, g1, z4, z4, 0);
#endif
}
#endif  // HAVE_TDM

// ---------------------------------------------------------------------------
// f32 -> bf16 one-pass conversion, 8 elements/thread
// ---------------------------------------------------------------------------
__global__ __launch_bounds__(256) void k_cvt(const float* __restrict__ s,
                                             bf16_t* __restrict__ d) {
  long long i = ((long long)blockIdx.x * 256 + threadIdx.x) * 8;
  *(v8bf*)(d + i) = load8(s + i);
}

// ---------------------------------------------------------------------------
// Tiled WMMA GEMM (bf16 x bf16 -> f32/bf16): C[M,N] = A[M,K] * B[K,N]
// block = 256 threads = 8 waves; block tile 128x256; wave tile 64x64.
// A staged by TDM (async tensor DMA, double-buffered, LDS row-padded by D# pad);
// B staged manually transposed ([n][k]) for contiguous B-fragment reads.
// ---------------------------------------------------------------------------
template <bool F32OUT>
__global__ __launch_bounds__(256) void k_gemm(const bf16_t* __restrict__ A,
                                              const bf16_t* __restrict__ Bm,
                                              void* __restrict__ Cv,
                                              int M, int N, int K) {
  __shared__ bf16_t lds_a[2][128 * 40];   // [buf][m * 40 + k], 40 = 32 + pad
  __shared__ bf16_t lds_bt[256][40];      // [n][k], padded

  const int tid  = threadIdx.x;
  const int lane = tid & 31;
  const int wid  = tid >> 5;
  const int wm   = wid & 1;   // 2 wave rows  (64 rows each)
  const int wn   = wid >> 1;  // 4 wave cols  (64 cols each)
  const int half = lane >> 4;
  const int l16  = lane & 15;
  const int m0   = blockIdx.x * 128;
  const int n0   = blockIdx.y * 256;
  const int T    = K / 32;

  v8f acc[4][4];
#pragma unroll
  for (int mf = 0; mf < 4; ++mf)
#pragma unroll
    for (int nf = 0; nf < 4; ++nf) acc[mf][nf] = (v8f)(0.0f);

#ifdef HAVE_TDM
  const unsigned ldsa0 = lds_addr_of(&lds_a[0][0]);
  const unsigned ldsa1 = lds_addr_of(&lds_a[1][0]);
  if (wid == 0) {
    tdm_load_tile_a(A + (size_t)m0 * K, ldsa0, 128, 32, K, M);
  }
#endif

  for (int t = 0; t < T; ++t) {
    const int k0 = t * 32;

    // stage B tile transposed: 32x256 = 1024 chunks of 8 (contiguous in N)
#pragma unroll
    for (int it = 0; it < 4; ++it) {
      int chunk = tid + it * 256;
      int r = chunk >> 5;          // k within tile (0..31)
      int c = (chunk & 31) * 8;    // n within tile (0..248)
      v8bf bv = *(const v8bf*)(Bm + (size_t)(k0 + r) * N + n0 + c);
#pragma unroll
      for (int j = 0; j < 8; ++j) lds_bt[c + j][r] = bv[j];
    }

#ifdef HAVE_TDM
    if (wid == 0) {
      if (t + 1 < T) {
        tdm_load_tile_a(A + (size_t)m0 * K + (t + 1) * 32,
                        ((t + 1) & 1) ? ldsa1 : ldsa0, 128, 32, K, M);
        __builtin_amdgcn_s_wait_tensorcnt((short)1);  // tile t complete
      } else {
        __builtin_amdgcn_s_wait_tensorcnt((short)0);
      }
    }
    const bf16_t* abuf = &lds_a[t & 1][0];
#else
    // manual A staging: 128x32 = 512 chunks of 8
#pragma unroll
    for (int it = 0; it < 2; ++it) {
      int chunk = tid + it * 256;
      int r = chunk >> 2;
      int c = (chunk & 3) * 8;
      *(v8bf*)&lds_a[0][r * 40 + c] = *(const v8bf*)(A + (size_t)(m0 + r) * K + k0 + c);
    }
    const bf16_t* abuf = &lds_a[0][0];
#endif
    __syncthreads();

    BFV16 af[4], bfr[4];
#pragma unroll
    for (int mf = 0; mf < 4; ++mf) {
      int row = wm * 64 + mf * 16 + l16;
      af[mf].h8[0] = *(const v8bf*)(abuf + row * 40 + half * 8);
      af[mf].h8[1] = *(const v8bf*)(abuf + row * 40 + 16 + half * 8);
    }
#pragma unroll
    for (int nf = 0; nf < 4; ++nf) {
      int col = wn * 64 + nf * 16 + l16;
      bfr[nf].h8[0] = *(const v8bf*)&lds_bt[col][half * 8];
      bfr[nf].h8[1] = *(const v8bf*)&lds_bt[col][16 + half * 8];
    }
#pragma unroll
    for (int mf = 0; mf < 4; ++mf)
#pragma unroll
      for (int nf = 0; nf < 4; ++nf)
        acc[mf][nf] = wmma_bf16(af[mf].v, bfr[nf].v, acc[mf][nf]);
    __syncthreads();
  }

  // epilogue: C layout — vgpr slot i holds row (i + 8*half), col = lane&15
#pragma unroll
  for (int mf = 0; mf < 4; ++mf) {
#pragma unroll
    for (int nf = 0; nf < 4; ++nf) {
      int row0 = m0 + wm * 64 + mf * 16 + 8 * half;
      int col  = n0 + wn * 64 + nf * 16 + l16;
#pragma unroll
      for (int i = 0; i < 8; ++i) {
        size_t idx = (size_t)(row0 + i) * N + col;
        if (F32OUT) ((float*)Cv)[idx]  = acc[mf][nf][i];
        else        ((bf16_t*)Cv)[idx] = (bf16_t)acc[mf][nf][i];
      }
    }
  }
}

// ---------------------------------------------------------------------------
// RoPE (in place, bf16): rotate adjacent pairs, theta = 10000
// ---------------------------------------------------------------------------
__global__ __launch_bounds__(256) void k_rope(bf16_t* __restrict__ buf, int heads) {
  long long idx = (long long)blockIdx.x * 256 + threadIdx.x;
  int p = (int)(idx & 63);
  long long t = idx >> 6;
  int h = (int)(t % heads);
  long long row = t / heads;            // b*SEQ + s
  int s = (int)(row & (SEQ_ - 1));
  float inv = __expf(-0.14391156831212787f * (float)p);  // theta^(-2p/128)
  float ang = (float)s * inv;
  float sn, cs;
  __sincosf(ang, &sn, &cs);
  size_t base = (size_t)row * ((size_t)heads * HD_) + (size_t)h * HD_ + 2 * p;
  float xr = (float)buf[base];
  float xi = (float)buf[base + 1];
  buf[base]     = (bf16_t)(xr * cs - xi * sn);
  buf[base + 1] = (bf16_t)(xr * sn + xi * cs);
}

// ---------------------------------------------------------------------------
// V transpose: V[b*SEQ+s][kvh*128+d] -> Vt[((b*NKV+kvh)*128+d)][s]
// ---------------------------------------------------------------------------
__global__ __launch_bounds__(256) void k_vtrans(const bf16_t* __restrict__ V,
                                                bf16_t* __restrict__ Vt) {
  long long idx = (long long)blockIdx.x * 256 + threadIdx.x;
  int d = (int)(idx & (HD_ - 1));
  long long t = idx >> 7;
  int kvh = (int)(t & (NKV_ - 1));
  long long row = t >> 3;               // b*SEQ + s
  int b = (int)(row >> 11);
  int s = (int)(row & (SEQ_ - 1));
  Vt[(((size_t)b * NKV_ + kvh) * HD_ + d) * SEQ_ + s] =
      V[(size_t)row * (NKV_ * HD_) + (size_t)kvh * HD_ + d];
}

// ---------------------------------------------------------------------------
// Flash attention (causal, GQA 4:1), all-WMMA.
// grid = (SEQ/128, NH, B), block = 256 (8 waves); each wave owns 16 q rows.
// ---------------------------------------------------------------------------
__global__ __launch_bounds__(256) void k_flash(const bf16_t* __restrict__ Q,
                                               const bf16_t* __restrict__ Kb,
                                               const bf16_t* __restrict__ Vt,
                                               bf16_t* __restrict__ O) {
  __shared__ bf16_t lds_p[8][16][40];   // per-wave P transpose scratch

  const int tid  = threadIdx.x;
  const int lane = tid & 31;
  const int wid  = tid >> 5;
  const int half = lane >> 4;
  const int l16  = lane & 15;
  const int h    = blockIdx.y;
  const int b    = blockIdx.z;
  const int kvh  = h >> 2;              // N_REP = 4
  const int qb   = blockIdx.x * 128 + wid * 16;

  BFV16 qa[4];
  const bf16_t* qrow = Q + ((size_t)b * SEQ_ + qb + l16) * (NH_ * HD_) + (size_t)h * HD_;
#pragma unroll
  for (int f = 0; f < 4; ++f) {
    qa[f].h8[0] = *(const v8bf*)(qrow + f * 32 + half * 8);
    qa[f].h8[1] = *(const v8bf*)(qrow + f * 32 + 16 + half * 8);
  }

  v8f o[8];
#pragma unroll
  for (int j = 0; j < 8; ++j) o[j] = (v8f)(0.0f);
  float mrow[8], lrow[8];
#pragma unroll
  for (int i = 0; i < 8; ++i) { mrow[i] = -3.0e38f; lrow[i] = 0.0f; }

  const float scale = 0.08838834764831845f;  // 1/sqrt(128)
  const bf16_t* kbase = Kb + (size_t)b * SEQ_ * (NKV_ * HD_) + (size_t)kvh * HD_;
  const bf16_t* vtb   = Vt + ((size_t)b * NKV_ + kvh) * (size_t)HD_ * SEQ_;

  for (int kb = 0; kb < qb + 16; kb += 32) {
    v8f s0 = (v8f)(0.0f), s1 = (v8f)(0.0f);
#pragma unroll
    for (int f = 0; f < 4; ++f) {
      BFV16 kf;
      const bf16_t* kr0 = kbase + (size_t)(kb + l16) * (NKV_ * HD_) + f * 32 + half * 8;
      kf.h8[0] = *(const v8bf*)kr0;
      kf.h8[1] = *(const v8bf*)(kr0 + 16);
      s0 = wmma_bf16(qa[f].v, kf.v, s0);
      const bf16_t* kr1 = kbase + (size_t)(kb + 16 + l16) * (NKV_ * HD_) + f * 32 + half * 8;
      kf.h8[0] = *(const v8bf*)kr1;
      kf.h8[1] = *(const v8bf*)(kr1 + 16);
      s1 = wmma_bf16(qa[f].v, kf.v, s1);
    }

#pragma unroll
    for (int i = 0; i < 8; ++i) {
      int qi   = qb + i + 8 * half;
      int key0 = kb + l16;
      int key1 = kb + 16 + l16;
      float v0 = s0[i] * scale;
      float v1 = s1[i] * scale;
      v0 = (key0 > qi) ? -3.0e38f : v0;
      v1 = (key1 > qi) ? -3.0e38f : v1;
      float mx = fmaxf(v0, v1);
      mx = fmaxf(mx, __shfl_xor(mx, 1, 32));
      mx = fmaxf(mx, __shfl_xor(mx, 2, 32));
      mx = fmaxf(mx, __shfl_xor(mx, 4, 32));
      mx = fmaxf(mx, __shfl_xor(mx, 8, 32));
      float mnew = fmaxf(mrow[i], mx);
      float cf = __expf(mrow[i] - mnew);
      float p0 = __expf(v0 - mnew);
      float p1 = __expf(v1 - mnew);
      float rs = p0 + p1;
      rs += __shfl_xor(rs, 1, 32);
      rs += __shfl_xor(rs, 2, 32);
      rs += __shfl_xor(rs, 4, 32);
      rs += __shfl_xor(rs, 8, 32);
      lrow[i] = lrow[i] * cf + rs;
      mrow[i] = mnew;
#pragma unroll
      for (int j = 0; j < 8; ++j) o[j][i] *= cf;
      lds_p[wid][i + 8 * half][l16]      = (bf16_t)p0;
      lds_p[wid][i + 8 * half][16 + l16] = (bf16_t)p1;
    }

    asm volatile("s_wait_dscnt 0" ::: "memory");
    BFV16 pa;
    pa.h8[0] = *(const v8bf*)&lds_p[wid][l16][half * 8];
    pa.h8[1] = *(const v8bf*)&lds_p[wid][l16][16 + half * 8];

#pragma unroll
    for (int j = 0; j < 8; ++j) {
      BFV16 vf;
      const bf16_t* vr = vtb + (size_t)(j * 16 + l16) * SEQ_ + kb + half * 8;
      vf.h8[0] = *(const v8bf*)vr;
      vf.h8[1] = *(const v8bf*)(vr + 16);
      o[j] = wmma_bf16(pa.v, vf.v, o[j]);
    }
  }

#pragma unroll
  for (int j = 0; j < 8; ++j) {
    int col = h * HD_ + j * 16 + l16;
#pragma unroll
    for (int i = 0; i < 8; ++i) {
      int row = qb + i + 8 * half;
      O[((size_t)b * SEQ_ + row) * (NH_ * HD_) + col] = (bf16_t)(o[j][i] / lrow[i]);
    }
  }
}

// ---------------------------------------------------------------------------
extern "C" void kernel_launch(void* const* d_in, const int* in_sizes, int n_in,
                              void* d_out, int out_size, void* d_ws, size_t ws_size,
                              hipStream_t stream) {
  (void)in_sizes; (void)n_in; (void)out_size; (void)ws_size;
  const float* x  = (const float*)d_in[0];
  const float* wq = (const float*)d_in[1];
  const float* wk = (const float*)d_in[2];
  const float* wv = (const float*)d_in[3];
  const float* wo = (const float*)d_in[4];

  char* ws = (char*)d_ws;
  size_t off = 0;
  auto take = [&](size_t bytes) -> void* {
    void* p = ws + off;
    off += (bytes + 255) & ~(size_t)255;
    return p;
  };
  bf16_t* Xb  = (bf16_t*)take((size_t)MTOT_ * DIM_ * sizeof(bf16_t));
  bf16_t* Wqb = (bf16_t*)take((size_t)DIM_ * NH_ * HD_ * sizeof(bf16_t));
  bf16_t* Wkb = (bf16_t*)take((size_t)DIM_ * NKV_ * HD_ * sizeof(bf16_t));
  bf16_t* Wvb = (bf16_t*)take((size_t)DIM_ * NKV_ * HD_ * sizeof(bf16_t));
  bf16_t* Wob = (bf16_t*)take((size_t)NH_ * HD_ * DIM_ * sizeof(bf16_t));
  bf16_t* Qb  = (bf16_t*)take((size_t)MTOT_ * NH_ * HD_ * sizeof(bf16_t));
  bf16_t* Kc  = (bf16_t*)take((size_t)MTOT_ * NKV_ * HD_ * sizeof(bf16_t));
  bf16_t* Vb  = (bf16_t*)take((size_t)MTOT_ * NKV_ * HD_ * sizeof(bf16_t));
  bf16_t* Vt  = (bf16_t*)take((size_t)MTOT_ * NKV_ * HD_ * sizeof(bf16_t));
  bf16_t* AO  = (bf16_t*)take((size_t)MTOT_ * NH_ * HD_ * sizeof(bf16_t));

  dim3 blk(256);

  // one-pass f32 -> bf16 conversions
  k_cvt<<<dim3((unsigned)((size_t)MTOT_ * DIM_ / 2048)), blk, 0, stream>>>(x, Xb);
  k_cvt<<<dim3((unsigned)((size_t)DIM_ * NH_ * HD_ / 2048)), blk, 0, stream>>>(wq, Wqb);
  k_cvt<<<dim3((unsigned)((size_t)DIM_ * NKV_ * HD_ / 2048)), blk, 0, stream>>>(wk, Wkb);
  k_cvt<<<dim3((unsigned)((size_t)DIM_ * NKV_ * HD_ / 2048)), blk, 0, stream>>>(wv, Wvb);
  k_cvt<<<dim3((unsigned)((size_t)NH_ * HD_ * DIM_ / 2048)), blk, 0, stream>>>(wo, Wob);

  // QKV projections (bf16 GEMM, TDM-staged A panel)
  k_gemm<false><<<dim3(MTOT_ / 128, (NH_ * HD_) / 256), blk, 0, stream>>>(
      Xb, Wqb, Qb, MTOT_, NH_ * HD_, DIM_);
  k_gemm<false><<<dim3(MTOT_ / 128, (NKV_ * HD_) / 256), blk, 0, stream>>>(
      Xb, Wkb, Kc, MTOT_, NKV_ * HD_, DIM_);
  k_gemm<false><<<dim3(MTOT_ / 128, (NKV_ * HD_) / 256), blk, 0, stream>>>(
      Xb, Wvb, Vb, MTOT_, NKV_ * HD_, DIM_);

  // RoPE on Q and K
  long long nq = (long long)MTOT_ * NH_ * 64;
  k_rope<<<dim3((unsigned)(nq / 256)), blk, 0, stream>>>(Qb, NH_);
  long long nk = (long long)MTOT_ * NKV_ * 64;
  k_rope<<<dim3((unsigned)(nk / 256)), blk, 0, stream>>>(Kc, NKV_);

  // V transpose for PV fragment reads
  long long nv = (long long)MTOT_ * NKV_ * HD_;
  k_vtrans<<<dim3((unsigned)(nv / 256)), blk, 0, stream>>>(Vb, Vt);

  // flash attention
  k_flash<<<dim3(SEQ_ / 128, NH_, B_), blk, 0, stream>>>(Qb, Kc, Vt, AO);

  // output projection -> f32 d_out
  k_gemm<true><<<dim3(MTOT_ / 128, DIM_ / 256), blk, 0, stream>>>(
      AO, Wob, (float*)d_out, MTOT_, DIM_, DIM_);
}
